// ConvLRUBlock_13821204758548
// MI455X (gfx1250) — matrix operand order repaired and checked by reading, not compiled, not run on hardware
//
#include <hip/hip_runtime.h>
#include <hip/hip_bf16.h>

typedef __attribute__((ext_vector_type(16))) _Float16 v16h;
typedef __attribute__((ext_vector_type(8)))  float    v8f;

#define C_CH   96
#define T_DIM  16
#define H_DIM  64
#define W_DIM  128
#define HW     (H_DIM * W_DIM)          // 8192
#define CSTR   (T_DIM * HW)             // 131072, channel stride in elements
#define CP     98                       // padded channel row (49 dwords -> conflict-free)
#define B_OFF  116480                   // halo bytes (9*66*98*2 = 116424) rounded up

static __device__ __forceinline__ int iclamp(int v, int lo, int hi) {
    return v < lo ? lo : (v > hi ? hi : v);
}

// ---------------------------------------------------------------------------
// Weight packing: OIDHW fp32 -> [tap][n][c] f16 (contiguous K per (tap,n))
// ---------------------------------------------------------------------------
__global__ __launch_bounds__(256) void pack_w_kernel(const float* __restrict__ w,
                                                     _Float16* __restrict__ o, int N) {
    int i = blockIdx.x * 256 + threadIdx.x;
    int total = 27 * N * C_CH;
    if (i >= total) return;
    int c   = i % C_CH;
    int n   = (i / C_CH) % N;
    int tap = i / (C_CH * N);
    o[((size_t)tap * N + n) * C_CH + c] = (_Float16)w[((size_t)n * C_CH + c) * 27 + tap];
}

// ---------------------------------------------------------------------------
// RMS norm over channels, write f16 activations
// ---------------------------------------------------------------------------
__global__ __launch_bounds__(256) void rms_kernel(const float* __restrict__ x,
                                                  const float* __restrict__ nw,
                                                  _Float16* __restrict__ h16, int total) {
    int i = blockIdx.x * 256 + threadIdx.x;
    if (i >= total) return;
    int w = i & 127, hh = (i >> 7) & 63, t = (i >> 13) & 15, b = i >> 17;
    size_t base = ((size_t)b * C_CH * T_DIM + t) * HW + hh * W_DIM + w;
    float s = 0.f;
    for (int c = 0; c < C_CH; ++c) {
        float v = x[base + (size_t)c * CSTR];
        s += v * v;
    }
    float r = rsqrtf(s * (1.0f / C_CH) + 1e-6f);
    for (int c = 0; c < C_CH; ++c)
        h16[base + (size_t)c * CSTR] = (_Float16)(x[base + (size_t)c * CSTR] * r * nw[c]);
}

// ---------------------------------------------------------------------------
// Async (ASYNCcnt-tracked) global->LDS dword copy, gfx1250 path.
// ---------------------------------------------------------------------------
static __device__ __forceinline__ void async_copy_b32(unsigned lds_byte_addr,
                                                      unsigned glb_byte_off,
                                                      const void* glb_base) {
    asm volatile("global_load_async_to_lds_b32 %0, %1, %2"
                 :: "v"(lds_byte_addr), "v"(glb_byte_off), "s"(glb_base)
                 : "memory");
}
static __device__ __forceinline__ void wait_async0() {
    asm volatile("s_wait_asynccnt 0x0" ::: "memory");
}

// ---------------------------------------------------------------------------
// Implicit-GEMM 3x3x3 conv via 27 tap-GEMMs, f16 WMMA 16x16x32, f32 accum.
// Block = 256 threads (8 waves, 4(M) x 2(N)); tile = 64 M-pixels x NW channels.
// Halo (3t x 3h x 66w x 96c, f16) resides in LDS for the whole block.
// B tiles double-buffered in LDS, filled with GLOBAL_LOAD_ASYNC_TO_LDS_B32.
// MODE 0: SiLU gate epilogue (NW=192, writes u f16). MODE 1: bias+residual (fp32 out).
// ---------------------------------------------------------------------------
template <int NW, int MODE>
__global__ __launch_bounds__(256) void conv_gemm_kernel(
    const _Float16* __restrict__ src,   // (B,C,T,H,W) f16
    const _Float16* __restrict__ pw,    // packed weights [27][NW][96] f16
    const float* __restrict__ bias,     // NW
    const float* __restrict__ xres,     // residual input (MODE 1)
    void* __restrict__ dst) {
    extern __shared__ char smem[];
    constexpr int BTILE_BYTES = NW * CP * 2;
    _Float16* halo = (_Float16*)smem;
    char*     bt0  = smem + B_OFF;
    char*     bt1  = smem + B_OFF + BTILE_BYTES;
    float*    stage = (float*)smem;     // aliases halo after compute

    const int bid = blockIdx.x;
    const int wseg = bid & 1, hh = (bid >> 1) & 63, t = (bid >> 7) & 15, b = bid >> 11;
    const int w0 = wseg * 64;
    const int tid = threadIdx.x;

    // ---- load halo once: [dtdh][wcol][c] with c padded to CP (transposing copy) ----
    for (int i = tid; i < 9 * C_CH * 66; i += 256) {
        int wi = i % 66;
        int c  = (i / 66) % C_CH;
        int dtdh = i / (66 * C_CH);
        int dt = dtdh / 3, dh = dtdh % 3;
        int tt = iclamp(t + dt - 1, 0, T_DIM - 1);
        int hy = iclamp(hh + dh - 1, 0, H_DIM - 1);
        int ww = (w0 + wi - 1 + W_DIM) & (W_DIM - 1);
        halo[(dtdh * 66 + wi) * CP + c] =
            src[((size_t)(b * C_CH + c) * T_DIM + tt) * HW + hy * W_DIM + ww];
    }

    // ---- async-stage B tile for tap 0 into buffer 0 ----
    {
        const void* gsrc = (const void*)pw;           // tap 0
        unsigned ldsbase = (unsigned)(uintptr_t)(const void*)bt0;
        for (int i = tid; i < NW * 48; i += 256) {
            int n = i / 48, cp2 = i % 48;
            async_copy_b32(ldsbase + (unsigned)(n * (CP * 2) + cp2 * 4),
                           (unsigned)(i * 4), gsrc);
        }
        wait_async0();
    }
    __syncthreads();

    const int wid = tid >> 5, lane = tid & 31;
    const int waveM = wid & 3, waveN = wid >> 2;
    const int lm = lane & 15, lhalf = lane >> 4;
    const int mB = waveM * 16;
    constexpr int NT = NW / 32;         // 16-wide N tiles per wave

    v8f acc[NT];
    for (int q = 0; q < NT; ++q)
        for (int r = 0; r < 8; ++r) acc[q][r] = 0.0f;

    for (int tap = 0; tap < 27; ++tap) {
        const char* cb = (tap & 1) ? bt1 : bt0;       // compute buffer
        char*       nb = (tap & 1) ? bt0 : bt1;       // fill buffer (tap+1)

        // issue async copies for the next tap; they complete while we compute
        if (tap + 1 < 27) {
            const void* gsrc = (const void*)(pw + (size_t)(tap + 1) * NW * C_CH);
            unsigned ldsbase = (unsigned)(uintptr_t)(const void*)nb;
            for (int i = tid; i < NW * 48; i += 256) {
                int n = i / 48, cp2 = i % 48;
                async_copy_b32(ldsbase + (unsigned)(n * (CP * 2) + cp2 * 4),
                               (unsigned)(i * 4), gsrc);
            }
            if (tap + 2 < 27)
                __builtin_prefetch(pw + (size_t)(tap + 2) * NW * C_CH + (tid & 63) * 32, 0, 0);
        }

        const int kd = tap / 9, kh = (tap / 3) % 3, kw = tap % 3;
        const int dtdh = kd * 3 + kh;
#pragma unroll
        for (int ks = 0; ks < 3; ++ks) {
            // A fragment: lanes 0-15 -> K 0-15, lanes 16-31 -> K 16-31 (matches B)
            union { v16h h; unsigned u[8]; } fa;
            {
                int wcol = mB + lm + kw;
                const unsigned* hp =
                    (const unsigned*)((const char*)halo + (dtdh * 66 + wcol) * (CP * 2));
                int c0d = (ks * 32 + lhalf * 16) >> 1;  // dword index into row
#pragma unroll
                for (int r = 0; r < 4; ++r) fa.u[r] = hp[c0d + r];
#pragma unroll
                for (int r = 0; r < 4; ++r) fa.u[4 + r] = hp[c0d + 4 + r];
            }
#pragma unroll
            for (int nt = 0; nt < NT; ++nt) {
                // B fragment: lanes 0-15 -> K 0-15, lanes 16-31 -> K 16-31, N = lane&15
                union { v16h h; unsigned u[8]; } fb;
                int n = waveN * (NW / 2) + nt * 16 + lm;
                const unsigned* bp = (const unsigned*)(cb + n * (CP * 2));
                int k0d = (ks * 32 + lhalf * 16) >> 1;
#pragma unroll
                for (int r = 0; r < 8; ++r) fb.u[r] = bp[k0d + r];
                acc[nt] = __builtin_amdgcn_wmma_f32_16x16x32_f16(
                    false, fa.h, false, fb.h, (short)0, acc[nt], false, false);
            }
        }

        wait_async0();       // next tap's tile resident
        __syncthreads();     // all waves done reading cb / writing nb
    }

    // reuse LDS (halo region) as fp32 stage for the epilogue
#pragma unroll
    for (int nt = 0; nt < NT; ++nt) {
        int n = waveN * (NW / 2) + nt * 16 + lm;
        int mbase = mB + lhalf * 8;
#pragma unroll
        for (int r = 0; r < 8; ++r) stage[(mbase + r) * NW + n] = acc[nt][r];
    }
    __syncthreads();

    size_t obase = ((size_t)b * C_CH * T_DIM + t) * HW + hh * W_DIM + w0;
    if (MODE == 0) {
        _Float16* u16 = (_Float16*)dst;
        for (int i = tid; i < C_CH * 64; i += 256) {
            int m = i & 63, c = i >> 6;
            float a = stage[m * NW + c] + bias[c];
            float g = stage[m * NW + C_CH + c] + bias[C_CH + c];
            float u = a * (1.0f / (1.0f + __expf(-a))) * g;
            u16[obase + (size_t)c * CSTR + m] = (_Float16)u;
        }
    } else {
        float* out = (float*)dst;
        for (int i = tid; i < C_CH * 64; i += 256) {
            int m = i & 63, c = i >> 6;
            size_t idx = obase + (size_t)c * CSTR + m;
            out[idx] = xres[idx] + stage[m * NW + c] + bias[c];
        }
    }
}

// ---------------------------------------------------------------------------
// LRU: complex linear recurrence over T=16 per (b,c,h,w) lane
// ---------------------------------------------------------------------------
__global__ __launch_bounds__(256) void lru_scan_kernel(
    const _Float16* __restrict__ u16, const float* __restrict__ nu_log,
    const float* __restrict__ theta_log, const float* __restrict__ c_re,
    const float* __restrict__ c_im, _Float16* __restrict__ y16, int total) {
    int i = blockIdx.x * 256 + threadIdx.x;
    if (i >= total) return;
    int hw = i & (HW - 1);
    int c  = (i >> 13) % C_CH;
    int b  = i / (C_CH * HW);
    float nu  = __expf(nu_log[c]);
    float th  = __expf(theta_log[c]);
    float mag = __expf(-nu);                       // |lambda|
    float lr = mag * __cosf(th), li = mag * __sinf(th);
    float gamma = sqrtf(fmaxf(0.f, 1.0f - mag * mag));
    float cre = c_re[c], cim = c_im[c];
    size_t base = (size_t)(b * C_CH + c) * T_DIM * HW + hw;
    float hr = 0.f, hi = 0.f;
#pragma unroll
    for (int t = 0; t < T_DIM; ++t) {
        float u  = (float)u16[base + (size_t)t * HW];
        float nr = lr * hr - li * hi + gamma * u;
        float ni = lr * hi + li * hr;
        hr = nr; hi = ni;
        y16[base + (size_t)t * HW] = (_Float16)(hr * cre + hi * cim);
    }
}

// ---------------------------------------------------------------------------
extern "C" void kernel_launch(void* const* d_in, const int* in_sizes, int n_in,
                              void* d_out, int out_size, void* d_ws, size_t ws_size,
                              hipStream_t stream) {
    const float* x          = (const float*)d_in[0];
    const float* norm_w     = (const float*)d_in[1];
    const float* conv_in_w  = (const float*)d_in[2];
    const float* conv_in_b  = (const float*)d_in[3];
    const float* nu_log     = (const float*)d_in[4];
    const float* theta_log  = (const float*)d_in[5];
    const float* c_re       = (const float*)d_in[6];
    const float* c_im       = (const float*)d_in[7];
    const float* conv_out_w = (const float*)d_in[8];
    const float* conv_out_b = (const float*)d_in[9];

    char* ws = (char*)d_ws;
    const size_t actBytes = (size_t)2 * C_CH * T_DIM * HW * sizeof(_Float16); // 50,331,648
    _Float16* h16   = (_Float16*)ws;                       // normalized acts; later aliased by y
    _Float16* u16   = (_Float16*)(ws + actBytes);          // gated acts
    _Float16* pwin  = (_Float16*)(ws + 2 * actBytes);      // 27*192*96 f16
    _Float16* pwout = (_Float16*)(ws + 2 * actBytes + 27 * 192 * C_CH * sizeof(_Float16));
    _Float16* y16   = h16;                                 // alias: h dead after conv_in

    dim3 blk(256);
    const int nCells = 2 * T_DIM * H_DIM * W_DIM;          // 262,144 (b,t,h,w)
    const int nScan  = 2 * C_CH * HW;                      // 1,572,864 (b,c,h,w)
    const int nGemmBlocks = 2 * T_DIM * H_DIM * (W_DIM / 64); // 4096

    pack_w_kernel<<<(27 * 192 * C_CH + 255) / 256, blk, 0, stream>>>(conv_in_w, pwin, 192);
    pack_w_kernel<<<(27 * C_CH * C_CH + 255) / 256, blk, 0, stream>>>(conv_out_w, pwout, C_CH);

    rms_kernel<<<(nCells + 255) / 256, blk, 0, stream>>>(x, norm_w, h16, nCells);

    // conv_in + SiLU gate: LDS = halo (116,480) + 2 x B tile (192*98*2)
    conv_gemm_kernel<192, 0><<<nGemmBlocks, blk, B_OFF + 2 * 192 * CP * 2, stream>>>(
        h16, pwin, conv_in_b, nullptr, (void*)u16);

    lru_scan_kernel<<<(nScan + 255) / 256, blk, 0, stream>>>(
        u16, nu_log, theta_log, c_re, c_im, y16, nScan);

    // conv_out + residual: LDS = halo + 2 x B tile (96*98*2)
    conv_gemm_kernel<96, 1><<<nGemmBlocks, blk, B_OFF + 2 * 96 * CP * 2, stream>>>(
        y16, pwout, conv_out_b, x, d_out);
}
